// TDBUCaptionBase_52226802320074
// MI455X (gfx1250) — compile-verified, matrix operand order checked by reading, not run on hardware
//
#include <hip/hip_runtime.h>
#include <hip/hip_bf16.h>
#include <math.h>

// ---------------------------------------------------------------------------
// Types for WMMA (gfx1250, wave32): v_wmma_f32_16x16x32_f16
// ---------------------------------------------------------------------------
typedef __attribute__((ext_vector_type(16))) _Float16 v16h;
typedef __attribute__((ext_vector_type(8)))  _Float16 v8h;
typedef __attribute__((ext_vector_type(8)))  float    v8f;

#define DEVFN __device__ __forceinline__

static constexpr int Bn = 1024;   // batch
static constexpr int Pp = 256;    // proposals
static constexpr int Ff = 128;    // obj feature dim
static constexpr int Hh = 512;    // hidden dim

#define WMMA_F16(a, b, c) \
  __builtin_amdgcn_wmma_f32_16x16x32_f16(false, (a), false, (b), (short)0, (c), false, false)

// Branch-free transcendentals (hardware v_tanh_f32 when available).
DEVFN float fast_tanh(float x) {
#if __has_builtin(__builtin_amdgcn_tanhf)
  return __builtin_amdgcn_tanhf(x);
#else
  float ax = __builtin_fabsf(x);
  float e  = __expf(-2.0f * ax);                                // v_exp_f32
  float t  = (1.0f - e) * __builtin_amdgcn_rcpf(1.0f + e);      // v_rcp_f32
  return copysignf(t, x);
#endif
}
DEVFN float fast_sigmoid(float x) {
  return __builtin_amdgcn_rcpf(1.0f + __expf(-x));
}

// Fragment loader for f16 row-major matrix (A: row=M, or W: row=N, K contiguous).
// Lane l holds row (l&15); lanes 0-15 carry K=k0+0..7 and k0+16..23,
// lanes 16-31 carry K=k0+8..15 and k0+24..31 (ISA 7.12.2, 16-bit A 16x32).
DEVFN v16h load_frag_f16(const _Float16* base, int ld, int row0, int k0, int lane) {
  int lr = lane & 15, hi = lane >> 4;
  const _Float16* p = base + (size_t)(row0 + lr) * ld + k0 + hi * 8;
  v8h lo = *(const v8h*)p;
  v8h hv = *(const v8h*)(p + 16);
  v16h r;
#pragma unroll
  for (int i = 0; i < 8; ++i) { r[i] = lo[i]; r[i + 8] = hv[i]; }
  return r;
}

// Same fragment, but source is f32 in memory (convert in-register).
DEVFN v16h load_frag_f32cvt(const float* rowp, int k0, int hi) {
  const float* p = rowp + k0 + hi * 8;
  v16h a;
#pragma unroll
  for (int i = 0; i < 8; ++i) { a[i] = (_Float16)p[i]; a[i + 8] = (_Float16)p[i + 16]; }
  return a;
}

// ---------------------------------------------------------------------------
// Generic WMMA GEMM:  C[M,N] = epi( A[M,K] * W[N,K]^T (+ bias[N]) )
// One 16x64 output tile per wave (4 accumulators): A-frag loaded once per
// k-step and reused 4x; 4 independent WMMA chains hide the RAW hazard.
// Requires N % 64 == 0 (true for all uses: 128/512/1536).
// EPI: 0 = f32 store, 1 = f32 store + bias, 2 = tanh -> f16, 3 = relu -> f16
// ---------------------------------------------------------------------------
template <int EPI>
__global__ void gemm_wmma_nt(const _Float16* __restrict__ A, int lda,
                             const _Float16* __restrict__ W, int ldw,
                             void* __restrict__ Cout, int ldc,
                             const float* __restrict__ bias,
                             int M, int N, int K) {
  int wave = (blockIdx.x * blockDim.x + threadIdx.x) >> 5;
  int lane = threadIdx.x & 31;
  int ntn = N >> 6;                        // 64-wide n groups
  int tiles = (M >> 4) * ntn;
  if (wave >= tiles) return;               // wave-uniform: EXEC stays all-ones
  int m0 = (wave / ntn) << 4;
  int n0 = (wave % ntn) << 6;

  v8f c0 = {}, c1 = {}, c2 = {}, c3 = {};
  for (int k0 = 0; k0 < K; k0 += 32) {
    v16h a  = load_frag_f16(A, lda, m0, k0, lane);
    v16h b0 = load_frag_f16(W, ldw, n0 +  0, k0, lane);
    v16h b1 = load_frag_f16(W, ldw, n0 + 16, k0, lane);
    v16h b2 = load_frag_f16(W, ldw, n0 + 32, k0, lane);
    v16h b3 = load_frag_f16(W, ldw, n0 + 48, k0, lane);
    c0 = WMMA_F16(a, b0, c0);
    c1 = WMMA_F16(a, b1, c1);
    c2 = WMMA_F16(a, b2, c2);
    c3 = WMMA_F16(a, b3, c3);
  }

  // C layout: element c[i] at lane l -> M = m0 + i + (l>=16 ? 8 : 0), N = n0 + (l&15)
  int lr    = lane & 15;
  int rbase = m0 + ((lane >> 4) << 3);
  v8f acc[4] = {c0, c1, c2, c3};
#pragma unroll
  for (int j = 0; j < 4; ++j) {
    int col = n0 + j * 16 + lr;
    float bs = (EPI == 1) ? bias[col] : 0.0f;
#pragma unroll
    for (int i = 0; i < 8; ++i) {
      size_t idx = (size_t)(rbase + i) * ldc + col;
      float v = acc[j][i];
      if (EPI == 0)      ((float*)Cout)[idx] = v;
      else if (EPI == 1) ((float*)Cout)[idx] = v + bs;
      else if (EPI == 2) ((_Float16*)Cout)[idx] = (_Float16)fast_tanh(v);
      else               ((_Float16*)Cout)[idx] = (_Float16)fmaxf(v, 0.0f);
    }
  }
}

// ---------------------------------------------------------------------------
// f32 -> f16 convert/pack with destination stride+offset (for K-concatenation)
// ---------------------------------------------------------------------------
__global__ void cvt_f32_f16(const float* __restrict__ src, _Float16* __restrict__ dst,
                            int rows, int cols, int dst_ld, int dst_off) {
  int total = rows * cols;
  for (int idx = blockIdx.x * blockDim.x + threadIdx.x; idx < total;
       idx += gridDim.x * blockDim.x) {
    int r = idx / cols, c = idx - r * cols;
    dst[(size_t)r * dst_ld + dst_off + c] = (_Float16)src[idx];
  }
}

// ---------------------------------------------------------------------------
// GRU elementwise: h' = (1-z)*n + z*h_prev, gates (r,z,n) in gi/gh (B x 3H).
// Optionally also emit f16 copy (strided) for downstream WMMA consumers.
// ---------------------------------------------------------------------------
__global__ void gru_elem(const float* __restrict__ gi, const float* __restrict__ gh,
                         const float* __restrict__ hprev, float* __restrict__ hout,
                         _Float16* __restrict__ h16out, int h16ld) {
  int idx = blockIdx.x * blockDim.x + threadIdx.x;
  if (idx >= Bn * Hh) return;
  int b = idx >> 9, h = idx & (Hh - 1);
  size_t o = (size_t)b * (3 * Hh);
  float r = fast_sigmoid(gi[o + h]      + gh[o + h]);
  float z = fast_sigmoid(gi[o + Hh + h] + gh[o + Hh + h]);
  float n = fast_tanh(gi[o + 2 * Hh + h] + r * gh[o + 2 * Hh + h]);
  float hn = (1.0f - z) * n + z * hprev[idx];
  hout[idx] = hn;
  if (h16out) h16out[(size_t)b * h16ld + h] = (_Float16)hn;
}

// ---------------------------------------------------------------------------
// Fused attention: one block per batch row b.
//   scores[p] = sum_h tanh( (obj[b] @ W_feat^T)[p,h] + hidproj[b,h] ) * watt[h]
//   masks    = softmax_p(scores)          -> d_out
//   attended = sum_p masks[p]*obj[b,p,:]  -> f16 into lang_cat cols [0,128)
// 8 waves; each wave owns 2 p-tiles (16 rows each); WMMA over 32 n-tiles, K=128.
// ---------------------------------------------------------------------------
__global__ void attention_fused(const float* __restrict__ obj,      // (B,P,F) f32
                                const _Float16* __restrict__ Wf,    // (H,F) f16
                                const float* __restrict__ hidproj,  // (B,H) f32
                                const float* __restrict__ watt,     // (H) f32
                                float* __restrict__ masks_out,      // (B,P)
                                _Float16* __restrict__ att_out,     // lang_cat base
                                int att_ld) {
  __shared__ float s_hid[Hh];
  __shared__ float s_watt[Hh];
  __shared__ float s_sc[Pp];
  __shared__ float s_red[256];

  int b = blockIdx.x;
  int t = threadIdx.x;
  const float* objb = obj + (size_t)b * Pp * Ff;

  for (int i = t; i < Hh; i += 256) {
    s_hid[i]  = hidproj[(size_t)b * Hh + i];
    s_watt[i] = watt[i];
  }
  __syncthreads();

  int wave = t >> 5, lane = t & 31, lr = lane & 15, hi = lane >> 4;

  for (int pt = wave; pt < Pp / 16; pt += 8) {     // 2 tiles per wave, uniform
    int p0 = pt << 4;
    const float* rowp = objb + (size_t)(p0 + lr) * Ff;
    v16h afr[4];
#pragma unroll
    for (int s = 0; s < 4; ++s) afr[s] = load_frag_f32cvt(rowp, s * 32, hi);

    float sacc[8];
#pragma unroll
    for (int i = 0; i < 8; ++i) sacc[i] = 0.0f;

    for (int nt = 0; nt < Hh / 16; ++nt) {
      int n0 = nt << 4;
      v16h b0 = load_frag_f16(Wf, Ff, n0,  0, lane);
      v16h b1 = load_frag_f16(Wf, Ff, n0, 32, lane);
      v16h b2 = load_frag_f16(Wf, Ff, n0, 64, lane);
      v16h b3 = load_frag_f16(Wf, Ff, n0, 96, lane);
      v8f c0 = {}, c1 = {};
      c0 = WMMA_F16(afr[0], b0, c0);
      c1 = WMMA_F16(afr[1], b1, c1);
      c0 = WMMA_F16(afr[2], b2, c0);
      c1 = WMMA_F16(afr[3], b3, c1);
      float hb = s_hid[n0 + lr];
      float wb = s_watt[n0 + lr];
#pragma unroll
      for (int i = 0; i < 8; ++i) sacc[i] += fast_tanh(c0[i] + c1[i] + hb) * wb;
    }
    // Reduce over the 16 h-columns held in each 16-lane half.
#pragma unroll
    for (int i = 0; i < 8; ++i) {
      float v = sacc[i];
      v += __shfl_xor(v, 1, 16);
      v += __shfl_xor(v, 2, 16);
      v += __shfl_xor(v, 4, 16);
      v += __shfl_xor(v, 8, 16);
      sacc[i] = v;
    }
    if (lr == 0) {
#pragma unroll
      for (int i = 0; i < 8; ++i) s_sc[p0 + hi * 8 + i] = sacc[i];
    }
  }
  __syncthreads();

  // Softmax over P=256 (one element per thread).
  float sc = s_sc[t];
  s_red[t] = sc;
  __syncthreads();
  for (int s = 128; s > 0; s >>= 1) {
    if (t < s) s_red[t] = fmaxf(s_red[t], s_red[t + s]);
    __syncthreads();
  }
  float mx = s_red[0];
  __syncthreads();
  float e = __expf(sc - mx);
  s_red[t] = e;
  __syncthreads();
  for (int s = 128; s > 0; s >>= 1) {
    if (t < s) s_red[t] += s_red[t + s];
    __syncthreads();
  }
  float prob = e * __builtin_amdgcn_rcpf(s_red[0]);
  __syncthreads();
  s_sc[t] = prob;
  masks_out[(size_t)b * Pp + t] = prob;
  __syncthreads();

  // attended[f] = sum_p prob[p] * obj[b,p,f]; two p-halves per f, then combine.
  int f  = t & (Ff - 1);
  int ph = t >> 7;                 // 0 or 1
  float acc = 0.0f;
  const float* op = objb + (size_t)ph * 128 * Ff + f;
  for (int p = 0; p < 128; ++p) acc += s_sc[ph * 128 + p] * op[(size_t)p * Ff];
  s_red[t] = acc;
  __syncthreads();
  if (t < Ff) {
    float v = s_red[t] + s_red[t + 128];
    att_out[(size_t)b * att_ld + t] = (_Float16)v;
  }
}

// ---------------------------------------------------------------------------
// Host orchestration
// ---------------------------------------------------------------------------
extern "C" void kernel_launch(void* const* d_in, const int* in_sizes, int n_in,
                              void* d_out, int out_size, void* d_ws, size_t ws_size,
                              hipStream_t stream) {
  (void)in_sizes; (void)n_in; (void)out_size; (void)ws_size;
  const float* step_input  = (const float*)d_in[0];   // (B,300)
  const float* target_feat = (const float*)d_in[1];   // (B,128)
  const float* obj_feats   = (const float*)d_in[2];   // (B,P,F)
  const float* hidden_1    = (const float*)d_in[3];   // (B,H)
  const float* hidden_2    = (const float*)d_in[4];   // (B,H)
  const float* W_td1 = (const float*)d_in[5];         // (128,512)
  const float* W_td2 = (const float*)d_in[6];         // (128,128)
  const float* W_td3 = (const float*)d_in[7];         // (128,300)
  const float* W_td  = (const float*)d_in[8];         // (128,128)
  const float* W_feat= (const float*)d_in[9];         // (512,128)
  const float* W_hidd= (const float*)d_in[10];        // (512,512)
  const float* W_att = (const float*)d_in[11];        // (1,512)
  const float* W_l1  = (const float*)d_in[12];        // (128,128)
  const float* W_l2  = (const float*)d_in[13];        // (128,512)
  const float* W_l   = (const float*)d_in[14];        // (128,128)
  const float* g1Wih = (const float*)d_in[15];        // (1536,128)
  const float* g1Whh = (const float*)d_in[16];        // (1536,512)
  const float* g1bih = (const float*)d_in[17];
  const float* g1bhh = (const float*)d_in[18];
  const float* g2Wih = (const float*)d_in[19];
  const float* g2Whh = (const float*)d_in[20];
  const float* g2bih = (const float*)d_in[21];
  const float* g2bhh = (const float*)d_in[22];

  float* out_h1    = (float*)d_out;                        // (B,H)
  float* out_h2    = out_h1 + (size_t)Bn * Hh;             // (B,H)
  float* out_masks = out_h2 + (size_t)Bn * Hh;             // (B,P)

  // Workspace carve-out (256B aligned so 16B vector loads are legal).
  char* ws = (char*)d_ws;
  size_t off = 0;
  auto carve = [&](size_t bytes) -> char* {
    char* p = ws + off;
    off = (off + bytes + 255) & ~(size_t)255;
    return p;
  };
  const int KCAT = 960;   // 300 + 512 + 128 padded to multiple of 32
  _Float16* A_cat    = (_Float16*)carve((size_t)Bn * KCAT * 2);
  _Float16* W_cat    = (_Float16*)carve((size_t)128 * KCAT * 2);
  _Float16* W_td16   = (_Float16*)carve((size_t)128 * 128 * 2);
  _Float16* Wih1_16  = (_Float16*)carve((size_t)1536 * 128 * 2);
  _Float16* Whh1_16  = (_Float16*)carve((size_t)1536 * 512 * 2);
  _Float16* W_feat16 = (_Float16*)carve((size_t)512 * 128 * 2);
  _Float16* W_hidd16 = (_Float16*)carve((size_t)512 * 512 * 2);
  _Float16* W_lcat   = (_Float16*)carve((size_t)128 * 640 * 2);
  _Float16* W_l16    = (_Float16*)carve((size_t)128 * 128 * 2);
  _Float16* Wih2_16  = (_Float16*)carve((size_t)1536 * 128 * 2);
  _Float16* Whh2_16  = (_Float16*)carve((size_t)1536 * 512 * 2);
  _Float16* td_t     = (_Float16*)carve((size_t)Bn * 128 * 2);
  _Float16* td16     = (_Float16*)carve((size_t)Bn * 128 * 2);
  _Float16* langt    = (_Float16*)carve((size_t)Bn * 128 * 2);
  _Float16* lang16   = (_Float16*)carve((size_t)Bn * 128 * 2);
  _Float16* h1p16    = (_Float16*)carve((size_t)Bn * Hh * 2);
  _Float16* h2p16    = (_Float16*)carve((size_t)Bn * Hh * 2);
  _Float16* lang_cat = (_Float16*)carve((size_t)Bn * 640 * 2);  // [attended | h1]
  float*    gi       = (float*)carve((size_t)Bn * 1536 * 4);
  float*    gh       = (float*)carve((size_t)Bn * 1536 * 4);
  float*    hidproj  = (float*)carve((size_t)Bn * Hh * 4);

  // Zero the padded concat buffers (pad columns must be 0).
  hipMemsetAsync(A_cat, 0, (size_t)Bn * KCAT * 2, stream);
  hipMemsetAsync(W_cat, 0, (size_t)128 * KCAT * 2, stream);

  auto cvt = [&](const float* s, _Float16* dst, int rows, int cols, int dld, int doff) {
    int total = rows * cols;
    int blocks = (total + 255) / 256;
    if (blocks > 4096) blocks = 4096;
    cvt_f32_f16<<<blocks, 256, 0, stream>>>(s, dst, rows, cols, dld, doff);
  };

  // K-concatenated fused input:  [step_input | hidden_2 | target_feat]
  cvt(step_input,  A_cat, Bn, 300, KCAT, 0);
  cvt(hidden_2,    A_cat, Bn, 512, KCAT, 300);
  cvt(target_feat, A_cat, Bn, 128, KCAT, 812);
  cvt(W_td3, W_cat, 128, 300, KCAT, 0);
  cvt(W_td1, W_cat, 128, 512, KCAT, 300);
  cvt(W_td2, W_cat, 128, 128, KCAT, 812);
  cvt(W_td,  W_td16, 128, 128, 128, 0);
  cvt(hidden_1, h1p16, Bn, 512, 512, 0);
  cvt(hidden_2, h2p16, Bn, 512, 512, 0);
  cvt(g1Wih, Wih1_16, 1536, 128, 128, 0);
  cvt(g1Whh, Whh1_16, 1536, 512, 512, 0);
  cvt(W_feat, W_feat16, 512, 128, 128, 0);
  cvt(W_hidd, W_hidd16, 512, 512, 512, 0);
  cvt(W_l1, W_lcat, 128, 128, 640, 0);
  cvt(W_l2, W_lcat, 128, 512, 640, 128);
  cvt(W_l,  W_l16, 128, 128, 128, 0);
  cvt(g2Wih, Wih2_16, 1536, 128, 128, 0);
  cvt(g2Whh, Whh2_16, 1536, 512, 512, 0);

#define GEMM(EPI, A, lda, Wt, ldw, C, ldc, bias, M, N, K)                      \
  do {                                                                         \
    int tiles_ = ((M) / 16) * ((N) / 64);                                      \
    int blocks_ = (tiles_ + 7) / 8;                                            \
    gemm_wmma_nt<EPI><<<blocks_, 256, 0, stream>>>((A), (lda), (Wt), (ldw),    \
                                                   (C), (ldc), (bias), (M),    \
                                                   (N), (K));                  \
  } while (0)

  // --- Top-down fusion + GRU1 ---
  GEMM(2, A_cat, KCAT, W_cat, KCAT, td_t, 128, nullptr, Bn, 128, KCAT);   // tanh
  GEMM(3, td_t, 128, W_td16, 128, td16, 128, nullptr, Bn, 128, 128);      // relu
  GEMM(1, td16, 128, Wih1_16, 128, gi, 1536, g1bih, Bn, 1536, 128);       // gi1
  GEMM(1, h1p16, 512, Whh1_16, 512, gh, 1536, g1bhh, Bn, 1536, 512);      // gh1
  gru_elem<<<(Bn * Hh) / 256, 256, 0, stream>>>(gi, gh, hidden_1, out_h1,
                                                lang_cat + 128, 640);     // h1 + f16

  // --- Attention ---
  GEMM(0, lang_cat + 128, 640, W_hidd16, 512, hidproj, 512, nullptr, Bn, Hh, Hh);
  attention_fused<<<Bn, 256, 0, stream>>>(obj_feats, W_feat16, hidproj, W_att,
                                          out_masks, lang_cat, 640);

  // --- Language fusion + GRU2 ---
  GEMM(2, lang_cat, 640, W_lcat, 640, langt, 128, nullptr, Bn, 128, 640); // tanh
  GEMM(3, langt, 128, W_l16, 128, lang16, 128, nullptr, Bn, 128, 128);    // relu
  GEMM(1, lang16, 128, Wih2_16, 128, gi, 1536, g2bih, Bn, 1536, 128);     // gi2
  GEMM(1, h2p16, 512, Whh2_16, 512, gh, 1536, g2bhh, Bn, 1536, 512);      // gh2
  gru_elem<<<(Bn * Hh) / 256, 256, 0, stream>>>(gi, gh, hidden_2, out_h2,
                                                nullptr, 0);
#undef GEMM
}
